// DockingHead_670014898683
// MI455X (gfx1250) — matrix operand order repaired and checked by reading, not compiled, not run on hardware
//
#include <hip/hip_runtime.h>
#include <hip/hip_bf16.h>

// DockingHead fused kernel for gfx1250 (MI455X).
// Exploits W1 block structure: h[b,i,j] = relu(glob_b@W1g + g_i@W1i + g_j@W1j + b1)
// -> two small WMMA GEMMs per graph-block + cheap per-pair combine, all in LDS.

typedef __attribute__((ext_vector_type(16))) _Float16 v16h;
typedef __attribute__((ext_vector_type(8)))  _Float16 v8h;
typedef __attribute__((ext_vector_type(4)))  _Float16 v4h;
typedef __attribute__((ext_vector_type(8)))  float    v8f;
typedef __attribute__((ext_vector_type(4)))  float    v4f;

#define GB   16    // graphs per workgroup
#define KP   128   // K stride (halfs) for f16 LDS tiles (rows 256B -> 16B aligned)
#define PQP  132   // padded N stride (floats) for PQ accumulator tile
#define NEGV (-1e9f)

// A-fragment (16x32 f16, M x K), from row-major [M][KP] f16 LDS tile.
// ISA layout: lanes 0-15 row M=lane, halves 0..7 = K 0..7, halves 8..15 = K 16..23;
// lanes 16-31 row M=lane-16, K 8..15 / 24..31.
__device__ __forceinline__ v16h load_a_frag(const _Float16* base, int m0, int koff) {
  const int lane = threadIdx.x & 31;
  const int row  = m0 + (lane & 15);
  const int kb   = koff + ((lane & 16) ? 8 : 0);
  const _Float16* p = base + row * KP + kb;
  v8h lo = *(const v8h*)(p);        // K kb .. kb+7
  v8h hi = *(const v8h*)(p + 16);   // K kb+16 .. kb+23
  return __builtin_shufflevector(lo, hi, 0,1,2,3,4,5,6,7,8,9,10,11,12,13,14,15);
}

// B-fragment (32x16 f16, K x N), from N-major (transposed) [N][KP] f16 LDS tile.
// ISA layout: lanes 0-15 col N=lane, halves = K 0..7 / 16..23; lanes 16-31 col N=lane-16, K 8..15 / 24..31.
__device__ __forceinline__ v16h load_b_frag(const _Float16* base, int n0, int koff) {
  const int lane = threadIdx.x & 31;
  const int row  = n0 + (lane & 15);
  const int kb   = koff + ((lane & 16) ? 8 : 0);
  const _Float16* p = base + row * KP + kb;
  v8h lo = *(const v8h*)(p);
  v8h hi = *(const v8h*)(p + 16);
  return __builtin_shufflevector(lo, hi, 0,1,2,3,4,5,6,7,8,9,10,11,12,13,14,15);
}

__global__ __launch_bounds__(256, 1)
void docking_head_kernel(const float* __restrict__ node,
                         const float* __restrict__ glob,
                         const int*   __restrict__ dock,
                         const float* __restrict__ W1,
                         const float* __restrict__ b1,
                         const float* __restrict__ W2,
                         const float* __restrict__ b2,
                         float* __restrict__ out) {
  __shared__ __align__(16) _Float16 actG [128 * KP];  // group nodes  [16 graphs * 8][128]
  __shared__ __align__(16) _Float16 actGl[ 16 * KP];  // globals      [16][128]
  __shared__ __align__(16) _Float16 wCat [128 * KP];  // [W1i|W1j]^T  N-major [128][128]
  __shared__ __align__(16) _Float16 wG   [ 64 * KP];  // W1g^T        N-major [64][128]
  __shared__ float PQ  [128 * PQP];                   // g @ [W1i|W1j]  f32
  __shared__ float Aout[ 16 * 68];                    // glob @ W1g     f32
  __shared__ float b1s[64], W2s[64];

  const int t    = threadIdx.x;
  const int wave = t >> 5;
  const int lane = t & 31;
  const int b0   = blockIdx.x * GB;

  // ---- stage group-node activations: 128 rows x 128 f32 -> f16 LDS ----
  #pragma unroll
  for (int it = 0; it < 16; ++it) {
    int q   = t + 256 * it;          // float4 index, 4096 total
    int row = q >> 5;                // 0..127  (= g*8 + i)
    int c4  = q & 31;
    int g = row >> 3, i = row & 7;
    v4f v = *(const v4f*)(node + (((size_t)(b0 + g)) * 64 + i) * 128 + c4 * 4);
    v4h h4; h4[0]=(_Float16)v[0]; h4[1]=(_Float16)v[1]; h4[2]=(_Float16)v[2]; h4[3]=(_Float16)v[3];
    *(v4h*)(actG + row * KP + c4 * 4) = h4;
  }
  // ---- stage globals: 16 rows x 128 f32 ----
  #pragma unroll
  for (int it = 0; it < 2; ++it) {
    int q   = t + 256 * it;          // 512 float4
    int row = q >> 5;                // 0..15
    int c4  = q & 31;
    v4f v = *(const v4f*)(glob + ((size_t)(b0 + row)) * 128 + c4 * 4);
    v4h h4; h4[0]=(_Float16)v[0]; h4[1]=(_Float16)v[1]; h4[2]=(_Float16)v[2]; h4[3]=(_Float16)v[3];
    *(v4h*)(actGl + row * KP + c4 * 4) = h4;
  }
  // ---- stage W1 (384x64 row-major) transposed into wG / wCat ----
  #pragma unroll 4
  for (int it = 0; it < 24; ++it) {
    int q    = t + 256 * it;         // 6144 float4
    int flat = q * 4;
    int f = flat >> 6;               // 0..383
    int h = flat & 63;
    v4f v = *(const v4f*)(W1 + flat);
    #pragma unroll
    for (int j = 0; j < 4; ++j) {
      _Float16 hv = (_Float16)v[j];
      int hh = h + j;
      if (f < 128)       wG  [hh * KP + f]            = hv;   // W1g^T
      else if (f < 256)  wCat[hh * KP + (f - 128)]    = hv;   // W1i^T -> cols 0..63
      else               wCat[(hh + 64) * KP + (f - 256)] = hv; // W1j^T -> cols 64..127
    }
  }
  if (t < 64) { b1s[t] = b1[t]; W2s[t] = W2[t]; }
  __syncthreads();

  // ---- WMMA GEMM 1: PQ[128x128] = actG[128x128] @ Wcat[128x128] ----
  {
    const int m0 = wave * 16;
    v16h af[4];
    #pragma unroll
    for (int s = 0; s < 4; ++s) af[s] = load_a_frag(actG, m0, 32 * s);
    #pragma unroll
    for (int nt = 0; nt < 8; ++nt) {
      v8f acc = {};
      #pragma unroll
      for (int s = 0; s < 4; ++s) {
        v16h bf = load_b_frag(wCat, nt * 16, 32 * s);
        acc = __builtin_amdgcn_wmma_f32_16x16x32_f16(false, af[s], false, bf,
                                                     (short)0, acc, false, false);
      }
      const int rbase = m0 + ((lane & 16) ? 8 : 0);
      const int col   = nt * 16 + (lane & 15);
      #pragma unroll
      for (int r = 0; r < 8; ++r) PQ[(rbase + r) * PQP + col] = acc[r];
    }
  }
  // ---- WMMA GEMM 2: Aout[16x64] = actGl[16x128] @ W1g[128x64] (waves 0..3) ----
  if (wave < 4) {
    v8f acc = {};
    #pragma unroll
    for (int s = 0; s < 4; ++s) {
      v16h af = load_a_frag(actGl, 0, 32 * s);
      v16h bf = load_b_frag(wG, wave * 16, 32 * s);
      acc = __builtin_amdgcn_wmma_f32_16x16x32_f16(false, af, false, bf,
                                                   (short)0, acc, false, false);
    }
    const int rbase = (lane & 16) ? 8 : 0;
    const int col   = wave * 16 + (lane & 15);
    #pragma unroll
    for (int r = 0; r < 8; ++r) Aout[(rbase + r) * 68 + col] = acc[r];
  }
  __syncthreads();

  // ---- combine: 16 graphs * 64 pairs; 4 scores per thread ----
  const float bias2 = b2[0];
  #pragma unroll
  for (int it = 0; it < 4; ++it) {
    int sidx = t + 256 * it;         // 0..1023
    int g    = sidx >> 6;
    int pair = sidx & 63;
    int i = pair >> 3, j = pair & 7;
    const float* Ag = Aout + g * 68;
    const float* P  = PQ + (g * 8 + i) * PQP;       // cols 0..63  (W1i part)
    const float* Q  = PQ + (g * 8 + j) * PQP + 64;  // cols 64..127 (W1j part)
    float s = bias2;
    #pragma unroll 8
    for (int h = 0; h < 64; ++h) {
      float hv = b1s[h] + Ag[h] + P[h] + Q[h];
      hv = fmaxf(hv, 0.0f);
      s = fmaf(W2s[h], hv, s);
    }
    const size_t bidx = (size_t)(b0 + g);
    const int m = dock[bidx * 64 + pair];
    out[bidx * 64 + pair] = (i == j || m == 0) ? NEGV : s;
  }
}

extern "C" void kernel_launch(void* const* d_in, const int* in_sizes, int n_in,
                              void* d_out, int out_size, void* d_ws, size_t ws_size,
                              hipStream_t stream) {
  const float* node = (const float*)d_in[0];   // node_features  [B*64, 128] f32
  const float* glob = (const float*)d_in[1];   // global_features [B, 128]   f32
  // d_in[2] group_mask_nodes, d_in[3] batch: structure is static, unused.
  const int*   dock = (const int*)d_in[4];     // dock_mask [B, 8, 8]
  const float* W1   = (const float*)d_in[5];   // [384, 64]
  const float* b1   = (const float*)d_in[6];   // [64]
  const float* W2   = (const float*)d_in[7];   // [64]
  const float* b2   = (const float*)d_in[8];   // [1]
  float* out = (float*)d_out;                  // [B, 64] f32

  const int B = in_sizes[1] / 128;
  dim3 grid(B / GB), block(256);
  hipLaunchKernelGGL(docking_head_kernel, grid, block, 0, stream,
                     node, glob, dock, W1, b1, W2, b2, out);
}
